// WideResNet_HyperDecisioNet_2_split_20126216749253
// MI455X (gfx1250) — compile-verified, matrix operand order checked by reading, not run on hardware
//
#include <hip/hip_runtime.h>
#include <cstdint>
#include <cstddef>

typedef _Float16 h16;
typedef __attribute__((ext_vector_type(16))) _Float16 v16h;
typedef __attribute__((ext_vector_type(8)))  _Float16 v8h;
typedef __attribute__((ext_vector_type(8)))  float    v8f;
typedef __attribute__((ext_vector_type(4)))  int      v4i;

#define EPS_BN 1e-5f

// ------------------------------------------------------------------
// gfx1250 async global->LDS helpers (builtin if available, else asm)
// builtin signature (probe-derived): arg0 = v4i* in AS(1) (global src),
// arg1 = v4i* in AS(3) (LDS dst), arg2 = imm offset, arg3 = imm cpol.
// ------------------------------------------------------------------
typedef __attribute__((address_space(1))) v4i* as1_v4i;
typedef __attribute__((address_space(3))) v4i* as3_v4i;

__device__ __forceinline__ void async_ld_b128(const h16* g, h16* l)
{
#if __has_builtin(__builtin_amdgcn_global_load_async_to_lds_b128)
  __builtin_amdgcn_global_load_async_to_lds_b128((as1_v4i)g, (as3_v4i)l, 0, 0);
#else
  unsigned loff = (unsigned)(size_t)l;                 // low 32 bits = LDS offset
  unsigned long long base = (unsigned long long)g & ~0xFULL;
  unsigned voff = (unsigned)((unsigned long long)g - base);
  asm volatile("global_load_async_to_lds_b128 %0, %1, %2 offset:0"
               :: "v"(loff), "v"(voff), "s"(base) : "memory");
#endif
}

__device__ __forceinline__ void wait_async0()
{
#if __has_builtin(__builtin_amdgcn_s_wait_asynccnt)
  __builtin_amdgcn_s_wait_asynccnt(0);
#else
  asm volatile("s_wait_asynccnt 0x0" ::: "memory");
#endif
}

// =====================================================================
// Elementwise / layout kernels
// =====================================================================

__global__ void nchw2nhwc_kernel(const float* __restrict__ x, h16* __restrict__ y,
                                 int Bn, int C, int H, int W)
{
  int t = blockIdx.x * blockDim.x + threadIdx.x;
  int tot = Bn * C * H * W;
  if (t >= tot) return;
  int w = t % W;
  int h = (t / W) % H;
  int c = (t / (W * H)) % C;
  int n = t / (W * H * C);
  y[(((size_t)n * H + h) * W + w) * C + c] = (h16)x[t];
}

// mode 0: plain f32->f16 cast.  mode 1: BN + ReLU then cast.
__global__ void prep_kernel(const float* __restrict__ x, h16* __restrict__ y,
                            const float* __restrict__ g, const float* __restrict__ b,
                            const float* __restrict__ mu, const float* __restrict__ var,
                            int N, int C, int mode)
{
  int t = blockIdx.x * blockDim.x + threadIdx.x;
  if (t >= N * C) return;
  float val = x[t];
  if (mode) {
    int c = t % C;
    val = (val - mu[c]) * (g[c] * rsqrtf(var[c] + EPS_BN)) + b[c];
    val = fmaxf(val, 0.0f);
  }
  y[t] = (h16)val;
}

// OIHW f32 weights -> wT[k][(r*R+s)*C + c] f16, rows padded with zeros to Kstride.
__global__ void convert_w_kernel(const float* __restrict__ w, h16* __restrict__ wT,
                                 int K, int C, int R, int Kdim, int Kstride)
{
  int t = blockIdx.x * blockDim.x + threadIdx.x;
  if (t >= K * Kstride) return;
  int k  = t / Kstride;
  int kk = t - k * Kstride;
  h16 val = (h16)0.0f;
  if (kk < Kdim) {
    int rs = kk / C;
    int c  = kk - rs * C;
    int rr = rs / R;
    int ss = rs - rr * R;
    val = (h16)w[(((size_t)k * C + c) * R + rr) * R + ss];
  }
  wT[t] = val;
}

__global__ void add_kernel(const float* __restrict__ a, const float* __restrict__ b,
                           float* __restrict__ o, int N)
{
  int t = blockIdx.x * blockDim.x + threadIdx.x;
  if (t < N) o[t] = a[t] + b[t];
}

// mean over H*W of NHWC tensor; writes out[b*ostride + ooff + c]
__global__ void gap_kernel(const float* __restrict__ x, float* __restrict__ out,
                           int HW, int C, int ostride, int ooff)
{
  int b = blockIdx.x;
  int c = threadIdx.x;
  if (c >= C) return;
  const float* p = x + (size_t)b * HW * C + c;
  float s = 0.0f;
  for (int i = 0; i < HW; ++i) s += p[(size_t)i * C];
  out[b * ostride + ooff + c] = s / (float)HW;
}

// final BN+ReLU fused into GAP
__global__ void gap_bnrelu_kernel(const float* __restrict__ x,
                                  const float* __restrict__ g, const float* __restrict__ b,
                                  const float* __restrict__ mu, const float* __restrict__ var,
                                  float* __restrict__ out, int HW, int C)
{
  int bi = blockIdx.x;
  int c  = threadIdx.x;
  if (c >= C) return;
  float sc = g[c] * rsqrtf(var[c] + EPS_BN);
  float mm = mu[c], bb = b[c];
  const float* p = x + (size_t)bi * HW * C + c;
  float s = 0.0f;
  for (int i = 0; i < HW; ++i) {
    float v = (p[(size_t)i * C] - mm) * sc + bb;
    s += fmaxf(v, 0.0f);
  }
  out[bi * C + c] = s / (float)HW;
}

// sig = feats @ w.T + b ; sb = sig[:,2] > 0 ; h = 0.5 + s/64
__global__ void selection_kernel(const float* __restrict__ feats, int F,
                                 const float* __restrict__ w, const float* __restrict__ bias,
                                 float* __restrict__ h0, float* __restrict__ h1, int Bn)
{
  int b = blockIdx.x * blockDim.x + threadIdx.x;
  if (b >= Bn) return;
  float s = bias[2];
  for (int f = 0; f < F; ++f) s += feats[b * F + f] * w[2 * F + f];
  float sb = (s > 0.0f) ? 1.0f : 0.0f;
  h0[b] = 0.5f + (1.0f - sb) * (1.0f / 64.0f);
  h1[b] = 0.5f + sb * (1.0f / 64.0f);
}

__global__ void fc_kernel(const float* __restrict__ xin, const float* __restrict__ w,
                          const float* __restrict__ bias, float* __restrict__ out,
                          int Bn, int F, int O)
{
  int t = blockIdx.x * blockDim.x + threadIdx.x;
  if (t >= Bn * O) return;
  int b = t / O;
  int o = t - b * O;
  float s = bias[o];
  for (int f = 0; f < F; ++f) s += xin[b * F + f] * w[o * F + f];
  out[t] = s;
}

// =====================================================================
// Implicit-GEMM conv via v_wmma_f32_16x16x32_f16, double-buffered LDS,
// async global->LDS for the weight tile (ASYNCcnt pipeline).
//   A[M x Kdim]: input patches (M = B*OH*OW), K ordered (r*R+s)*Cin + c
//   B[Kdim x Cout]: weights as wT[Cout][Kstride] (zero padded rows)
//   out = GEMM  (+ addsrc, optionally scaled per-sample by ps[b])
// Block: 128 threads = 4 waves, 64x64 tile, each wave 32x32 (2x2 wmma).
// =====================================================================

#define LDS_PITCH 40   // halfwords; 80B row pitch -> conflict-free b128 LDS reads

__global__ __launch_bounds__(128, 2)
void conv_wmma_kernel(const h16* __restrict__ in, const h16* __restrict__ wT,
                      float* __restrict__ out,
                      const float* __restrict__ addsrc, const float* __restrict__ ps,
                      int Bn, int IH, int IW, int Cin,
                      int OH, int OW, int Cout,
                      int R, int pad, int stride, int Kstride)
{
  __shared__ h16 As[2][64][LDS_PITCH];
  __shared__ h16 Bs[2][64][LDS_PITCH];

  const int Kdim = R * R * Cin;
  const int OHW  = OH * OW;
  const int Mtot = Bn * OHW;
  const int m0 = blockIdx.x * 64;
  const int n0 = blockIdx.y * 64;

  const int tid = threadIdx.x;
  const int row = tid >> 1;              // 0..63: tile row this thread loads
  const int seg = (tid & 1) << 4;        // 0 or 16: K segment

  // decode output pixel for the A row we load (constant over K loop)
  const int m = m0 + row;
  const bool mval = (m < Mtot);
  int n_s = 0, oh = 0, ow = 0;
  if (mval) { n_s = m / OHW; int rem = m - n_s * OHW; oh = rem / OW; ow = rem - oh * OW; }
  const int ih0 = oh * stride - pad;
  const int iw0 = ow * stride - pad;

  const int nrow   = n0 + row;
  const int nclamp = (nrow < Cout) ? nrow : 0;         // clamped: junk cols masked at store
  const h16* wrowp = wT + (size_t)nclamp * Kstride;

  const int wave = tid >> 5;
  const int lane = tid & 31;
  const int wm = (wave >> 1) << 5;       // wave M offset in tile (0/32)
  const int wn = (wave & 1) << 5;        // wave N offset in tile (0/32)
  const int half = lane >> 4;
  const int l16  = lane & 15;

  v8f acc[2][2] = {};

  const int nCh = Kstride >> 5;
  const bool fastA = ((Cin & 31) == 0);

  union Frag { v16h v; v8h h[2]; };

  // ---- tile fill: A via manual gather (zero padding), B via async DMA ----
  auto loadA = [&](int ch, int buf) {
    union { v8h h2[2]; h16 e[16]; } tmp;
#pragma unroll
    for (int j = 0; j < 16; ++j) tmp.e[j] = (h16)0.0f;
    const int kc = (ch << 5) + seg;
    if (mval) {
      if (fastA) {
        int rs = kc / Cin;
        int c0 = kc - rs * Cin;
        int rr = rs / R;
        int ss = rs - rr * R;
        int ih = ih0 + rr, iw = iw0 + ss;
        if ((unsigned)ih < (unsigned)IH && (unsigned)iw < (unsigned)IW) {
          const v8h* p = (const v8h*)(in + (((size_t)n_s * IH + ih) * IW + iw) * Cin + c0);
          tmp.h2[0] = p[0];
          tmp.h2[1] = p[1];
        }
      } else {
#pragma unroll
        for (int j = 0; j < 16; ++j) {
          int k = kc + j;
          if (k < Kdim) {
            int rs = k / Cin;
            int c  = k - rs * Cin;
            int rr = rs / R;
            int ss = rs - rr * R;
            int ih = ih0 + rr, iw = iw0 + ss;
            if ((unsigned)ih < (unsigned)IH && (unsigned)iw < (unsigned)IW)
              tmp.e[j] = in[(((size_t)n_s * IH + ih) * IW + iw) * Cin + c];
          }
        }
      }
    }
    *(v8h*)&As[buf][row][seg]     = tmp.h2[0];
    *(v8h*)&As[buf][row][seg + 8] = tmp.h2[1];
  };
  auto loadB = [&](int ch, int buf) {
    const h16* g0 = wrowp + (ch << 5) + seg;
    h16* l0 = &Bs[buf][row][seg];
    async_ld_b128(g0,     l0);        // 16B
    async_ld_b128(g0 + 8, l0 + 8);    // 16B
  };

  // prologue: fill buffer 0
  loadA(0, 0);
  loadB(0, 0);
  wait_async0();
  __syncthreads();

  for (int ch = 0; ch < nCh; ++ch) {
    const int cur = ch & 1;
    // kick off next chunk's fills into the other buffer (overlaps with WMMA)
    if (ch + 1 < nCh) {
      loadA(ch + 1, cur ^ 1);
      loadB(ch + 1, cur ^ 1);
    }

    // ---------------- fragments + WMMA on current buffer ----------------
    Frag af[2], bf[2];
#pragma unroll
    for (int i = 0; i < 2; ++i) {
      int rA = wm + (i << 4) + l16;
      af[i].h[0] = *(const v8h*)&As[cur][rA][half << 3];
      af[i].h[1] = *(const v8h*)&As[cur][rA][16 + (half << 3)];
    }
#pragma unroll
    for (int j = 0; j < 2; ++j) {
      int rB = wn + (j << 4) + l16;
      bf[j].h[0] = *(const v8h*)&Bs[cur][rB][half << 3];
      bf[j].h[1] = *(const v8h*)&Bs[cur][rB][16 + (half << 3)];
    }
#pragma unroll
    for (int i = 0; i < 2; ++i)
#pragma unroll
      for (int j = 0; j < 2; ++j)
        acc[i][j] = __builtin_amdgcn_wmma_f32_16x16x32_f16(
            false, af[i].v, false, bf[j].v, (short)0, acc[i][j], false, false);

    // next buffer must be complete before the next iteration reads it
    wait_async0();
    __syncthreads();
  }

  // ---------------- epilogue: (+ addsrc [* ps]) store ----------------
#pragma unroll
  for (int i = 0; i < 2; ++i) {
#pragma unroll
    for (int j = 0; j < 2; ++j) {
      int gn = n0 + wn + (j << 4) + l16;
      if (gn >= Cout) continue;
#pragma unroll
      for (int vv = 0; vv < 8; ++vv) {
        int gm = m0 + wm + (i << 4) + (half << 3) + vv;
        if (gm >= Mtot) continue;
        float val = acc[i][j][vv];
        if (addsrc) {
          float a = addsrc[(size_t)gm * Cout + gn];
          if (ps) a *= ps[gm / OHW];
          val += a;
        }
        out[(size_t)gm * Cout + gn] = val;
      }
    }
  }
}

// =====================================================================
// Host orchestration
// =====================================================================

struct BNp { const float *g, *b, *m, *v; };
struct BasicP { BNp bn1; const float* c1; BNp bn2; const float* c2; const float* sc; };
struct HyperP { BNp bn1; const float *hw1, *hb1; BNp bn2; const float *hw2, *hb2; const float* sc; };

extern "C" void kernel_launch(void* const* d_in, const int* in_sizes, int n_in,
                              void* d_out, int out_size, void* d_ws, size_t ws_size,
                              hipStream_t stream)
{
  (void)in_sizes; (void)n_in; (void)out_size; (void)ws_size;
  const int Bn = 32;

  // ---------------- parse inputs in setup_inputs() order ----------------
  int idx = 0;
  auto nxt = [&]() { return (const float*)d_in[idx++]; };
  auto getBN = [&]() { BNp r; r.g = nxt(); r.b = nxt(); r.m = nxt(); r.v = nxt(); return r; };

  const float* x_in    = nxt();
  const float* conv1_w = nxt();

  BasicP L1[4];
  L1[0].bn1 = getBN(); L1[0].c1 = nxt(); L1[0].bn2 = getBN(); L1[0].c2 = nxt(); L1[0].sc = nxt();
  for (int i = 1; i < 4; ++i) {
    L1[i].bn1 = getBN(); L1[i].c1 = nxt(); L1[i].bn2 = getBN(); L1[i].c2 = nxt(); L1[i].sc = nullptr;
  }

  HyperP H2;
  H2.bn1 = getBN(); H2.hw1 = nxt(); H2.hb1 = nxt(); H2.bn2 = getBN(); H2.hw2 = nxt(); H2.hb2 = nxt(); H2.sc = nxt();
  BasicP L2[3];
  for (int i = 0; i < 3; ++i) {
    L2[i].bn1 = getBN(); L2[i].c1 = nxt(); L2[i].bn2 = getBN(); L2[i].c2 = nxt(); L2[i].sc = nullptr;
  }

  HyperP H3;
  H3.bn1 = getBN(); H3.hw1 = nxt(); H3.hb1 = nxt(); H3.bn2 = getBN(); H3.hw2 = nxt(); H3.hb2 = nxt(); H3.sc = nxt();
  BasicP L3[3];
  for (int i = 0; i < 3; ++i) {
    L3[i].bn1 = getBN(); L3[i].c1 = nxt(); L3[i].bn2 = getBN(); L3[i].c2 = nxt(); L3[i].sc = nullptr;
  }

  BNp bnF = getBN();
  const float* fc_w  = nxt();
  const float* fc_b  = nxt();
  const float* sel1w = nxt();
  const float* sel1b = nxt();
  const float* sel2w = nxt();
  const float* sel2b = nxt();

  // ---------------- workspace layout ----------------
  const size_t MAXACT = (size_t)32 * 32 * 32 * 160;   // 5,242,880 elements
  const size_t PATHB  = (size_t)32 * 16 * 16 * 160;   // 1,310,720 elements
  char* cur_ws = (char*)d_ws;
  auto alloc = [&](size_t bytes) {
    void* r = (void*)cur_ws;
    cur_ws += (bytes + 255) & ~(size_t)255;
    return r;
  };
  float* A  = (float*)alloc(MAXACT * 4);
  float* Bb = (float*)alloc(MAXACT * 4);
  float* Cc = (float*)alloc(MAXACT * 4);
  float* D  = (float*)alloc(MAXACT * 4);
  float* E  = (float*)alloc(PATHB * 4);
  float* F  = (float*)alloc(PATHB * 4);
  h16* Ha = (h16*)alloc(MAXACT * 2);
  h16* Hb = (h16*)alloc(MAXACT * 2);
  h16* Hc = (h16*)alloc(MAXACT * 2);
  h16* WT = (h16*)alloc((size_t)320 * 2880 * 2);
  float* feats1 = (float*)alloc(32 * 160 * 4);
  float* feats2 = (float*)alloc(32 * 320 * 4);
  float* h0v    = (float*)alloc(32 * 4);
  float* h1v    = (float*)alloc(32 * 4);
  float* pooled = (float*)alloc(32 * 320 * 4);

  // ---------------- launch helpers ----------------
  int curKstride = 32;
  auto do_w = [&](const float* wsrc, int K, int C, int R) {
    int Kdim = R * R * C;
    int Kst  = (Kdim + 31) & ~31;
    int tot  = K * Kst;
    convert_w_kernel<<<(tot + 255) / 256, 256, 0, stream>>>(wsrc, WT, K, C, R, Kdim, Kst);
    curKstride = Kst;
  };
  auto do_conv = [&](const h16* inp, float* outp, const float* adds, const float* psb,
                     int IH, int Cin, int Cout, int R, int pad, int stride) {
    int IW = IH;
    int OHt = (IH + 2 * pad - R) / stride + 1;
    int OWt = OHt;
    int M = Bn * OHt * OWt;
    dim3 g((M + 63) / 64, (Cout + 63) / 64);
    conv_wmma_kernel<<<g, 128, 0, stream>>>(inp, WT, outp, adds, psb,
                                            Bn, IH, IW, Cin, OHt, OWt, Cout,
                                            R, pad, stride, curKstride);
  };
  auto do_prep = [&](const float* src, h16* dst, const BNp* bn, int N, int C) {
    int tot = N * C;
    prep_kernel<<<(tot + 255) / 256, 256, 0, stream>>>(
        src, dst, bn ? bn->g : nullptr, bn ? bn->b : nullptr,
        bn ? bn->m : nullptr, bn ? bn->v : nullptr, N, C, bn ? 1 : 0);
  };
  auto run_basic = [&](const BasicP& p, const float* curb, float* t1, float* outbuf,
                       int HH, int C) {
    int N = Bn * HH * HH;
    do_prep(curb, Ha, &p.bn1, N, C);
    do_w(p.c1, C, C, 3);
    do_conv(Ha, t1, nullptr, nullptr, HH, C, C, 3, 1, 1);
    do_prep(t1, Hb, &p.bn2, N, C);
    do_w(p.c2, C, C, 3);
    do_conv(Hb, outbuf, curb, nullptr, HH, C, C, 3, 1, 1);
  };
  auto run_hyper = [&](const HyperP& p, const h16* Hp, const h16* Hx, const float* hps,
                       float* tmp, float* mid, float* outbuf, int IH, int Cin, int Cout) {
    int OHs = IH / 2;
    do_w(p.hw1, Cout, Cin, 3);
    do_conv(Hp, tmp, nullptr, nullptr, IH, Cin, Cout, 3, 1, 2);
    do_w(p.hb1, Cout, Cin, 3);
    do_conv(Hp, mid, tmp, hps, IH, Cin, Cout, 3, 1, 2);
    do_prep(mid, Hc, &p.bn2, Bn * OHs * OHs, Cout);
    do_w(p.hw2, Cout, Cout, 3);
    do_conv(Hc, tmp, nullptr, nullptr, OHs, Cout, Cout, 3, 1, 1);
    do_w(p.hb2, Cout, Cout, 3);
    do_conv(Hc, mid, tmp, hps, OHs, Cout, Cout, 3, 1, 1);
    do_w(p.sc, Cout, Cin, 1);
    do_conv(Hx, outbuf, mid, nullptr, IH, Cin, Cout, 1, 0, 2);
  };

  // ---------------- network ----------------
  // stem: conv1 (3->16, 3x3, s1 p1) on 32x32
  {
    int tot = Bn * 3 * 32 * 32;
    nchw2nhwc_kernel<<<(tot + 255) / 256, 256, 0, stream>>>(x_in, Ha, Bn, 3, 32, 32);
  }
  do_w(conv1_w, 16, 3, 3);
  do_conv(Ha, A, nullptr, nullptr, 32, 3, 16, 3, 1, 1);

  // layer1 block0 (16 -> 160, shortcut)
  do_prep(A, Ha, &L1[0].bn1, Bn * 1024, 16);
  do_w(L1[0].c1, 160, 16, 3);
  do_conv(Ha, Bb, nullptr, nullptr, 32, 16, 160, 3, 1, 1);
  do_prep(Bb, Hb, &L1[0].bn2, Bn * 1024, 160);
  do_prep(A, Hc, nullptr, Bn * 1024, 16);
  do_w(L1[0].sc, 160, 16, 1);
  do_conv(Hc, Cc, nullptr, nullptr, 32, 16, 160, 1, 0, 1);
  do_w(L1[0].c2, 160, 160, 3);
  do_conv(Hb, D, Cc, nullptr, 32, 160, 160, 3, 1, 1);
  // layer1 blocks 1..3 (160 -> 160)
  run_basic(L1[1], D, Bb, Cc, 32, 160);
  run_basic(L1[2], Cc, Bb, D, 32, 160);
  run_basic(L1[3], D, Bb, Cc, 32, 160);   // layer1 out = Cc (32x32x160)

  // selection 1 + stash gap(x_prev) for selection 2
  gap_kernel<<<Bn, 160, 0, stream>>>(Cc, feats1, 1024, 160, 160, 0);
  gap_kernel<<<Bn, 160, 0, stream>>>(Cc, feats2, 1024, 160, 320, 160);
  selection_kernel<<<1, 32, 0, stream>>>(feats1, 160, sel1w, sel1b, h0v, h1v, Bn);

  // layer2: hyper (160->160, s2) x2 paths + 3 basics each
  do_prep(Cc, Ha, &H2.bn1, Bn * 1024, 160);
  do_prep(Cc, Hb, nullptr, Bn * 1024, 160);
  run_hyper(H2, Ha, Hb, h0v, Bb, D, E, 32, 160, 160);
  run_hyper(H2, Ha, Hb, h1v, Bb, D, F, 32, 160, 160);
  run_basic(L2[0], E, Bb, D, 16, 160);
  run_basic(L2[1], D, Bb, E, 16, 160);
  run_basic(L2[2], E, Bb, D, 16, 160);    // x0 = D
  run_basic(L2[0], F, Bb, Cc, 16, 160);
  run_basic(L2[1], Cc, Bb, F, 16, 160);
  run_basic(L2[2], F, Bb, Cc, 16, 160);   // x1 = Cc
  {
    int N = Bn * 256 * 160;
    add_kernel<<<(N + 255) / 256, 256, 0, stream>>>(D, Cc, A, N);  // x = A (16x16x160)
  }
  gap_kernel<<<Bn, 160, 0, stream>>>(A, feats2, 256, 160, 320, 0);
  selection_kernel<<<1, 32, 0, stream>>>(feats2, 320, sel2w, sel2b, h0v, h1v, Bn);

  // layer3: hyper (160->320, s2) x2 paths + 3 basics each
  do_prep(A, Ha, &H3.bn1, Bn * 256, 160);
  do_prep(A, Hb, nullptr, Bn * 256, 160);
  run_hyper(H3, Ha, Hb, h0v, Bb, D, E, 16, 160, 320);
  run_hyper(H3, Ha, Hb, h1v, Bb, D, F, 16, 160, 320);
  run_basic(L3[0], E, Bb, D, 8, 320);
  run_basic(L3[1], D, Bb, E, 8, 320);
  run_basic(L3[2], E, Bb, D, 8, 320);     // x0 = D
  run_basic(L3[0], F, Bb, Cc, 8, 320);
  run_basic(L3[1], Cc, Bb, F, 8, 320);
  run_basic(L3[2], F, Bb, Cc, 8, 320);    // x1 = Cc
  {
    int N = Bn * 64 * 320;
    add_kernel<<<(N + 255) / 256, 256, 0, stream>>>(D, Cc, A, N);  // x = A (8x8x320)
  }

  // head: BN+ReLU -> GAP -> FC
  gap_bnrelu_kernel<<<Bn, 320, 0, stream>>>(A, bnF.g, bnF.b, bnF.m, bnF.v, pooled, 64, 320);
  {
    int tot = Bn * 100;
    fc_kernel<<<(tot + 127) / 128, 128, 0, stream>>>(pooled, fc_w, fc_b, (float*)d_out, Bn, 320, 100);
  }
}
